// MultiHeadAttention_16492674417134
// MI455X (gfx1250) — matrix operand order repaired and checked by reading, not compiled, not run on hardware
//
#include <hip/hip_runtime.h>
#include <hip/hip_bf16.h>
#include <stdint.h>

// ---------------------------------------------------------------------------
// MHA for MI455X (gfx1250): bf16 WMMA everywhere, flash-attention fusion,
// TDM tensor_load_to_lds for K tiles, async global->LDS double-buffered GEMMs.
// B=2, S=2048, D=2048, H=16, HD=128
// ---------------------------------------------------------------------------

#define B_  2
#define S_  2048
#define D_  2048
#define H_  16
#define HD_ 128

typedef __attribute__((ext_vector_type(16))) __bf16 bf16x16;
typedef __attribute__((ext_vector_type(8)))  __bf16 bf16x8;
typedef __attribute__((ext_vector_type(8)))  float  floatx8;
typedef __attribute__((ext_vector_type(4)))  unsigned uint32x4;
typedef __attribute__((ext_vector_type(8)))  int      int32x8;
typedef __attribute__((ext_vector_type(4)))  int      int32x4;

#define WMMA_BF16(a, b, c) \
  __builtin_amdgcn_wmma_f32_16x16x32_bf16(false, (a), false, (b), (short)0, (c), false, false)

// Build a 16-element bf16 fragment from two 16-byte chunks (ISA A/B layouts).
__device__ __forceinline__ bf16x16 ld_frag(const __bf16* p0, const __bf16* p1) {
  bf16x8 lo = *(const bf16x8*)p0;
  bf16x8 hi = *(const bf16x8*)p1;
  bf16x16 r;
#pragma unroll
  for (int i = 0; i < 8; ++i) { r[i] = lo[i]; r[i + 8] = hi[i]; }
  return r;
}

// CDNA5 async copy: global -> LDS, 16 bytes per lane, tracked by ASYNCcnt.
__device__ __forceinline__ void async_copy_b128(void* lds_ptr, const void* gptr) {
  unsigned l = (unsigned)(uintptr_t)lds_ptr;
  unsigned long long g = (unsigned long long)(uintptr_t)gptr;
  asm volatile("global_load_async_to_lds_b128 %0, %1, off"
               :: "v"(l), "v"(g) : "memory");
}
__device__ __forceinline__ void wait_async0() {
  asm volatile("s_wait_asynccnt 0" ::: "memory");
}

// TDM: load a 2D tile [tile_rows x row_elems] of bf16 (contiguous rows,
// row stride = row_stride elems) from global into LDS at lds_ptr.
// One issue per wave (EXEC ignored); tracked by TENSORcnt.
__device__ __forceinline__ void tdm_load_2d_bf16(void* lds_ptr, const void* gptr,
                                                 unsigned row_elems, unsigned tensor_rows,
                                                 unsigned tile_rows, unsigned row_stride) {
  unsigned lds = (unsigned)(uintptr_t)lds_ptr;
  unsigned long long ga = (unsigned long long)(uintptr_t)gptr;
  // D# group 0: count=1 | lds_addr | global_addr[56:0] | type=2 (ISA 8.3)
  uint32x4 g0 = { 1u, lds, (unsigned)ga,
                  (unsigned)((ga >> 32) & 0x01FFFFFFu) | (2u << 30) };
  // D# group 1 (ISA 8.4): data_size=1 (2B); tensor_dim0/1; tile_dim0/1; stride0.
  int32x8 g1 = { (int)(1u << 16),                 // data_size=2B, no mask/pad/iter
                 (int)(row_elems << 16),          // tensor_dim0[15:0] in [31:16]
                 (int)(tensor_rows << 16),        // dim0[31:16]=0 | tensor_dim1[15:0]
                 (int)(row_elems << 16),          // dim1[31:16]=0 | tile_dim0
                 (int)tile_rows,                  // tile_dim1 | tile_dim2=0
                 (int)row_stride,                 // tensor_dim0_stride[31:0]
                 0, 0 };                          // stride0[47:32], dim1_stride
  int32x4 gz4 = { 0, 0, 0, 0 };                   // 2D tensor: groups 2/3 unused
  int32x8 gz8 = { 0, 0, 0, 0, 0, 0, 0, 0 };
  __builtin_amdgcn_tensor_load_to_lds(g0, g1, gz4, gz4, gz8, 0);
}

// ---------------------------------------------------------------------------
// fp32 -> bf16 convert
// ---------------------------------------------------------------------------
__global__ void cvt_f32_bf16(const float* __restrict__ in, __bf16* __restrict__ out, int n) {
  int i = blockIdx.x * blockDim.x + threadIdx.x;
  int stride = gridDim.x * blockDim.x;
  for (; i < n; i += stride) out[i] = (__bf16)in[i];
}

// in: [rows][cols] fp32 row-major  ->  out: [cols][rows] bf16 (transposed)
__global__ void transpose_cvt(const float* __restrict__ in, __bf16* __restrict__ out,
                              int rows, int cols) {
  long i = (long)blockIdx.x * blockDim.x + threadIdx.x;
  long total = (long)rows * cols;
  long stride = (long)gridDim.x * blockDim.x;
  for (; i < total; i += stride) {
    long n = i / rows;
    long k = i % rows;
    out[i] = (__bf16)in[k * (long)cols + n];
  }
}

// ---------------------------------------------------------------------------
// QKV projection: [4096 x 2048] @ [2048 x 6144] (+bias) -> Q,K,V [B,H,S,HD] bf16
// ---------------------------------------------------------------------------
__global__ __launch_bounds__(256) void gemm_qkv_kernel(
    const __bf16* __restrict__ A,    // [B*S][D] bf16
    const __bf16* __restrict__ Bt,   // [3D][D]  bf16 (w_qkv transposed)
    const float*  __restrict__ bias, // [3D]
    __bf16* __restrict__ Qo, __bf16* __restrict__ Ko, __bf16* __restrict__ Vo) {
  const int K = D_;
  __shared__ __align__(16) __bf16 As[2][128 * 40];  // row-major M x K, stride 40
  __shared__ __align__(16) __bf16 Bs[2][128 * 40];  // row-major N x K, stride 40

  const int tid  = threadIdx.x;
  const int wave = tid >> 5;
  const int lane = tid & 31;
  const int lr   = lane & 15;
  const int hi   = lane >> 4;
  const int m_base = blockIdx.x * 128;
  const int n_base = blockIdx.y * 128;

  // Each thread stages 2 x 16B chunks for A and for B per k-step.
  const int r0 = tid >> 2,          c0 = (tid & 3) << 3;
  const int r1 = (tid + 256) >> 2,  c1 = ((tid + 256) & 3) << 3;

  auto issue_tile = [&](int k0, __bf16* Ad, __bf16* Bd) {
    async_copy_b128(&Ad[r0 * 40 + c0], &A[(size_t)(m_base + r0) * K + k0 + c0]);
    async_copy_b128(&Bd[r0 * 40 + c0], &Bt[(size_t)(n_base + r0) * K + k0 + c0]);
    async_copy_b128(&Ad[r1 * 40 + c1], &A[(size_t)(m_base + r1) * K + k0 + c1]);
    async_copy_b128(&Bd[r1 * 40 + c1], &Bt[(size_t)(n_base + r1) * K + k0 + c1]);
  };

  floatx8 acc[8];
#pragma unroll
  for (int j = 0; j < 8; ++j)
#pragma unroll
    for (int e = 0; e < 8; ++e) acc[j][e] = 0.f;

  __bf16 *Ac = As[0], *Bc = Bs[0], *An = As[1], *Bn = Bs[1];
  const int nk = K / 32;
  issue_tile(0, Ac, Bc);
  for (int i = 0; i < nk; ++i) {
    wait_async0();       // this wave's async writes have landed
    __syncthreads();     // everyone's writes landed / prior reads retired
    if (i + 1 < nk) issue_tile((i + 1) * 32, An, Bn);

    bf16x16 af = ld_frag(&Ac[(wave * 16 + lr) * 40 + hi * 8],
                         &Ac[(wave * 16 + lr) * 40 + hi * 8 + 16]);
#pragma unroll
    for (int j = 0; j < 8; ++j) {
      bf16x16 bfg = ld_frag(&Bc[(j * 16 + lr) * 40 + hi * 8],
                            &Bc[(j * 16 + lr) * 40 + hi * 8 + 16]);
      acc[j] = WMMA_BF16(af, bfg, acc[j]);
    }
    __bf16* t;
    t = Ac; Ac = An; An = t;
    t = Bc; Bc = Bn; Bn = t;
  }

  const float qscale = 0.08838834764831845f;  // HD^-0.5, folded into Q
#pragma unroll
  for (int j = 0; j < 8; ++j) {
#pragma unroll
    for (int r = 0; r < 8; ++r) {
      int gm = m_base + wave * 16 + hi * 8 + r;  // row in [B*S)
      int gn = n_base + j * 16 + lr;             // col in [3D)
      float v = acc[j][r] + bias[gn];
      int b = gm >> 11, s = gm & 2047;
      int trip = gn >> 11, rem = gn & 2047, h = rem >> 7, hd = rem & 127;
      size_t off = (((size_t)(b * H_ + h)) * S_ + s) * HD_ + hd;
      if (trip == 0)      Qo[off] = (__bf16)(v * qscale);
      else if (trip == 1) Ko[off] = (__bf16)v;
      else                Vo[off] = (__bf16)v;
    }
  }
}

// ---------------------------------------------------------------------------
// Flash attention: one WG per (b*h, 128-query tile). 8 waves x 16 rows.
// K tiles staged via TDM (TENSORcnt); V staged with on-the-fly transpose.
// ---------------------------------------------------------------------------
__global__ __launch_bounds__(256) void attn_kernel(
    const __bf16* __restrict__ Qb, const __bf16* __restrict__ Kb,
    const __bf16* __restrict__ Vb, __bf16* __restrict__ Ob) {
  __shared__ __align__(16) __bf16 Ks[64 * 128];      // [key][dim]
  __shared__ __align__(16) __bf16 Vt[128 * 72];      // [dim][key] (transposed)
  __shared__ __align__(16) __bf16 Ps[8 * 16 * 72];   // per-wave P staging

  const int tid  = threadIdx.x;
  const int wave = tid >> 5;
  const int lane = tid & 31;
  const int lr   = lane & 15;
  const int hi   = lane >> 4;
  const int bh = blockIdx.x;        // b*H + h
  const int q0 = blockIdx.y * 128;

  const __bf16* Qh = Qb + (size_t)bh * S_ * HD_;
  const __bf16* Kh = Kb + (size_t)bh * S_ * HD_;
  const __bf16* Vh = Vb + (size_t)bh * S_ * HD_;

  // Q fragments: 16 rows x HD=128, 4 chunks of K=32, kept in registers.
  bf16x16 qf[4];
  {
    const __bf16* qrow = Qh + (size_t)(q0 + wave * 16 + lr) * HD_;
#pragma unroll
    for (int c = 0; c < 4; ++c)
      qf[c] = ld_frag(qrow + c * 32 + hi * 8, qrow + c * 32 + hi * 8 + 16);
  }

  floatx8 oacc[8];
#pragma unroll
  for (int j = 0; j < 8; ++j)
#pragma unroll
    for (int e = 0; e < 8; ++e) oacc[j][e] = 0.f;
  float m_i[8], l_i[8];
#pragma unroll
  for (int r = 0; r < 8; ++r) { m_i[r] = -1e30f; l_i[r] = 0.f; }

  const int nkt = (q0 + 128) / 64;  // causal: only key tiles <= query tile
  for (int kt = 0; kt < nkt; ++kt) {
    const int kb = kt * 64;
    __syncthreads();  // previous tile fully consumed by all waves
    // K tile (64x128 bf16, contiguous rows): single TDM descriptor, wave 0.
    if (wave == 0)
      tdm_load_2d_bf16(&Ks[0], Kh + (size_t)kb * HD_,
                       /*row_elems=*/HD_, /*tensor_rows=*/S_,
                       /*tile_rows=*/64, /*row_stride=*/HD_);
    // V tile via transpose through VGPRs (TDM cannot transpose).
#pragma unroll
    for (int c = 0; c < 4; ++c) {
      int idx = tid + c * 256;        // 0..1023 chunks of 8 bf16
      int row = idx >> 4;             // 0..63
      int col = (idx & 15) << 3;      // 0..120
      uint4 vv = *(const uint4*)&Vh[(size_t)(kb + row) * HD_ + col];
      const __bf16* pv = (const __bf16*)&vv;
#pragma unroll
      for (int e = 0; e < 8; ++e) Vt[(col + e) * 72 + row] = pv[e];
    }
    if (wave == 0) __builtin_amdgcn_s_wait_tensorcnt(0);
    __syncthreads();

    // scores = Q . K^T : 16 x 64 per wave (4 n-subtiles x 4 K-chunks)
    floatx8 sacc[4];
#pragma unroll
    for (int jn = 0; jn < 4; ++jn)
#pragma unroll
      for (int e = 0; e < 8; ++e) sacc[jn][e] = 0.f;
#pragma unroll
    for (int jn = 0; jn < 4; ++jn) {
#pragma unroll
      for (int c = 0; c < 4; ++c) {
        bf16x16 bfg = ld_frag(&Ks[(jn * 16 + lr) * 128 + c * 32 + hi * 8],
                              &Ks[(jn * 16 + lr) * 128 + c * 32 + hi * 8 + 16]);
        sacc[jn] = WMMA_BF16(qf[c], bfg, sacc[jn]);
      }
    }

    // Causal mask + online softmax (per row; rows split across lane halves).
#pragma unroll
    for (int r = 0; r < 8; ++r) {
      const int rowg = q0 + wave * 16 + hi * 8 + r;
      float rmax = -1e30f;
#pragma unroll
      for (int jn = 0; jn < 4; ++jn) {
        int colg = kb + jn * 16 + lr;
        float sv = (colg > rowg) ? -1e30f : sacc[jn][r];
        sacc[jn][r] = sv;
        rmax = fmaxf(rmax, sv);
      }
#pragma unroll
      for (int m = 1; m < 16; m <<= 1)
        rmax = fmaxf(rmax, __shfl_xor(rmax, m, 32));
      float mnew  = fmaxf(m_i[r], rmax);
      float alpha = __expf(m_i[r] - mnew);
      m_i[r] = mnew;
      float psum = 0.f;
#pragma unroll
      for (int jn = 0; jn < 4; ++jn) {
        float p = __expf(sacc[jn][r] - mnew);
        sacc[jn][r] = p;
        psum += p;
      }
#pragma unroll
      for (int m = 1; m < 16; m <<= 1)
        psum += __shfl_xor(psum, m, 32);
      l_i[r] = l_i[r] * alpha + psum;
#pragma unroll
      for (int jd = 0; jd < 8; ++jd) oacc[jd][r] *= alpha;
    }

    // Re-shape P (C layout) -> A layout via per-wave LDS staging.
    __bf16* Pw = &Ps[wave * 16 * 72];
#pragma unroll
    for (int jn = 0; jn < 4; ++jn)
#pragma unroll
      for (int r = 0; r < 8; ++r)
        Pw[(hi * 8 + r) * 72 + jn * 16 + lr] = (__bf16)sacc[jn][r];

    // O += P . V : 2 key-chunks x 8 dim-subtiles
#pragma unroll
    for (int c2 = 0; c2 < 2; ++c2) {
      bf16x16 af = ld_frag(&Pw[lr * 72 + c2 * 32 + hi * 8],
                           &Pw[lr * 72 + c2 * 32 + hi * 8 + 16]);
#pragma unroll
      for (int jd = 0; jd < 8; ++jd) {
        bf16x16 bfg = ld_frag(&Vt[(jd * 16 + lr) * 72 + c2 * 32 + hi * 8],
                              &Vt[(jd * 16 + lr) * 72 + c2 * 32 + hi * 8 + 16]);
        oacc[jd] = WMMA_BF16(af, bfg, oacc[jd]);
      }
    }
  }

  // Normalize and write attn output in [B,S,D] bf16 (ready for out-proj GEMM).
  const int b = bh >> 4, h = bh & 15;
#pragma unroll
  for (int jd = 0; jd < 8; ++jd) {
#pragma unroll
    for (int r = 0; r < 8; ++r) {
      int srow = q0 + wave * 16 + hi * 8 + r;
      size_t row = (size_t)b * S_ + srow;
      int col = h * HD_ + jd * 16 + lr;
      Ob[row * D_ + col] = (__bf16)(oacc[jd][r] / l_i[r]);
    }
  }
}

// ---------------------------------------------------------------------------
// Output projection: [4096 x 2048] @ [2048 x 2048] (+bias) -> fp32 d_out
// ---------------------------------------------------------------------------
__global__ __launch_bounds__(256) void gemm_out_kernel(
    const __bf16* __restrict__ A,    // [B*S][D] bf16 (attn)
    const __bf16* __restrict__ Bt,   // [D][D] bf16 (w_out transposed)
    const float*  __restrict__ bias, // [D]
    float* __restrict__ out) {
  const int K = D_;
  __shared__ __align__(16) __bf16 As[2][128 * 40];
  __shared__ __align__(16) __bf16 Bs[2][128 * 40];

  const int tid  = threadIdx.x;
  const int wave = tid >> 5;
  const int lane = tid & 31;
  const int lr   = lane & 15;
  const int hi   = lane >> 4;
  const int m_base = blockIdx.x * 128;
  const int n_base = blockIdx.y * 128;

  const int r0 = tid >> 2,          c0 = (tid & 3) << 3;
  const int r1 = (tid + 256) >> 2,  c1 = ((tid + 256) & 3) << 3;

  auto issue_tile = [&](int k0, __bf16* Ad, __bf16* Bd) {
    async_copy_b128(&Ad[r0 * 40 + c0], &A[(size_t)(m_base + r0) * K + k0 + c0]);
    async_copy_b128(&Bd[r0 * 40 + c0], &Bt[(size_t)(n_base + r0) * K + k0 + c0]);
    async_copy_b128(&Ad[r1 * 40 + c1], &A[(size_t)(m_base + r1) * K + k0 + c1]);
    async_copy_b128(&Bd[r1 * 40 + c1], &Bt[(size_t)(n_base + r1) * K + k0 + c1]);
  };

  floatx8 acc[8];
#pragma unroll
  for (int j = 0; j < 8; ++j)
#pragma unroll
    for (int e = 0; e < 8; ++e) acc[j][e] = 0.f;

  __bf16 *Ac = As[0], *Bc = Bs[0], *An = As[1], *Bn = Bs[1];
  const int nk = K / 32;
  issue_tile(0, Ac, Bc);
  for (int i = 0; i < nk; ++i) {
    wait_async0();
    __syncthreads();
    if (i + 1 < nk) issue_tile((i + 1) * 32, An, Bn);

    bf16x16 af = ld_frag(&Ac[(wave * 16 + lr) * 40 + hi * 8],
                         &Ac[(wave * 16 + lr) * 40 + hi * 8 + 16]);
#pragma unroll
    for (int j = 0; j < 8; ++j) {
      bf16x16 bfg = ld_frag(&Bc[(j * 16 + lr) * 40 + hi * 8],
                            &Bc[(j * 16 + lr) * 40 + hi * 8 + 16]);
      acc[j] = WMMA_BF16(af, bfg, acc[j]);
    }
    __bf16* t;
    t = Ac; Ac = An; An = t;
    t = Bc; Bc = Bn; Bn = t;
  }

#pragma unroll
  for (int j = 0; j < 8; ++j) {
#pragma unroll
    for (int r = 0; r < 8; ++r) {
      int gm = m_base + wave * 16 + hi * 8 + r;
      int gn = n_base + j * 16 + lr;
      out[(size_t)gm * D_ + gn] = acc[j][r] + bias[gn];
    }
  }
}

// ---------------------------------------------------------------------------
// Host-side orchestration (all on `stream`; workspace carved from d_ws).
// ---------------------------------------------------------------------------
extern "C" void kernel_launch(void* const* d_in, const int* in_sizes, int n_in,
                              void* d_out, int out_size, void* d_ws, size_t ws_size,
                              hipStream_t stream) {
  (void)in_sizes; (void)n_in; (void)out_size; (void)ws_size;
  const float* x     = (const float*)d_in[0];
  // d_in[1] = attention_mask: causal by construction, applied analytically.
  const float* w_qkv = (const float*)d_in[2];
  const float* b_qkv = (const float*)d_in[3];
  const float* w_out = (const float*)d_in[4];
  const float* b_out = (const float*)d_in[5];
  float* out = (float*)d_out;

  char* ws = (char*)d_ws;
  size_t off = 0;
  auto carve = [&](size_t elems) {
    __bf16* p = (__bf16*)(ws + off);
    off += ((elems * sizeof(__bf16) + 255) & ~(size_t)255);
    return p;
  };
  __bf16* x_bf  = carve((size_t)B_ * S_ * D_);      // x in bf16
  __bf16* wqkvT = carve((size_t)3 * D_ * D_);       // w_qkv^T bf16 [3D][D]
  __bf16* woutT = carve((size_t)D_ * D_);           // w_out^T bf16 [D][D]
  __bf16* Qb    = carve((size_t)B_ * S_ * D_);      // [B,H,S,HD] bf16 (pre-scaled)
  __bf16* Kb    = carve((size_t)B_ * S_ * D_);
  __bf16* Vb    = carve((size_t)B_ * S_ * D_);
  __bf16* attn  = carve((size_t)B_ * S_ * D_);      // [B,S,D] bf16

  cvt_f32_bf16<<<4096, 256, 0, stream>>>(x, x_bf, B_ * S_ * D_);
  transpose_cvt<<<4096, 256, 0, stream>>>(w_qkv, wqkvT, D_, 3 * D_);
  transpose_cvt<<<4096, 256, 0, stream>>>(w_out, woutT, D_, D_);

  gemm_qkv_kernel<<<dim3((B_ * S_) / 128, (3 * D_) / 128), 256, 0, stream>>>(
      x_bf, wqkvT, b_qkv, Qb, Kb, Vb);

  attn_kernel<<<dim3(B_ * H_, S_ / 128), 256, 0, stream>>>(Qb, Kb, Vb, attn);

  gemm_out_kernel<<<dim3((B_ * S_) / 128, D_ / 128), 256, 0, stream>>>(
      attn, woutT, b_out, out);
}
